// RNN_40776419508632
// MI455X (gfx1250) — matrix operand order repaired and checked by reading, not compiled, not run on hardware
//
#include <hip/hip_runtime.h>
#include <math.h>

typedef __attribute__((ext_vector_type(16))) _Float16 v16h;
typedef __attribute__((ext_vector_type(8)))  _Float16 v8h;
typedef __attribute__((ext_vector_type(4)))  _Float16 v4h;
typedef __attribute__((ext_vector_type(8)))  float    v8f;
typedef __attribute__((ext_vector_type(4)))  float    v4f;

#define DD     512           // feature dim (in == hidden == out)
#define MT     64            // tokens per block
#define STRIDE 528           // LDS row stride in halves (512 + 16 pad)
#define NTOK   (8 * 2048)    // B*S tokens
#define YTOT   ((size_t)NTOK * DD)

#if defined(__has_builtin) && __has_builtin(__builtin_amdgcn_tanhf)
#  define TANHF(x) __builtin_amdgcn_tanhf(x)
#else
#  define TANHF(x) tanhf(x)
#endif

union FragH { v16h v; v8h h[2]; };

// ---- Convert the three f32 weight matrices to f16 in workspace -------------
__global__ void cvt_w_f16(const float* __restrict__ W0,
                          const float* __restrict__ W1,
                          const float* __restrict__ Wt,
                          _Float16* __restrict__ o) {
  int i4 = blockIdx.x * blockDim.x + threadIdx.x;   // one float4 per matrix per thread
  const v4f* s0 = (const v4f*)W0;
  const v4f* s1 = (const v4f*)W1;
  const v4f* s2 = (const v4f*)Wt;
  v4h* d0 = (v4h*)o;
  v4h* d1 = (v4h*)(o + DD * DD);
  v4h* d2 = (v4h*)(o + 2 * DD * DD);
  v4f a = s0[i4], b = s1[i4], c = s2[i4];
  v4h ha, hb, hc;
#pragma unroll
  for (int j = 0; j < 4; ++j) {
    ha[j] = (_Float16)a[j];
    hb[j] = (_Float16)b[j];
    hc[j] = (_Float16)c[j];
  }
  d0[i4] = ha; d1[i4] = hb; d2[i4] = hc;
}

// ---- Fused 3-layer MLP: x -> tanh -> tanh -> y, intermediates in LDS -------
__global__ void __launch_bounds__(256, 2)
mlp3_wmma(const float* __restrict__ x,
          const _Float16* __restrict__ W16,
          const float* __restrict__ b0,
          const float* __restrict__ b1,
          const float* __restrict__ bt,
          float* __restrict__ out) {
  extern __shared__ _Float16 lds[];
  _Float16* A = lds;                 // current layer input  [MT x STRIDE] f16
  _Float16* H = lds + MT * STRIDE;   // current layer output [MT x STRIDE] f16

  const int tid  = threadIdx.x;
  const int base = blockIdx.x * MT;  // first token of this band

  // Phase 0: stage x band into LDS as f16 (vectorized f32x4 -> f16x4)
  for (int i4 = tid; i4 < MT * (DD / 4); i4 += 256) {
    int row = i4 >> 7;
    int c4  = (i4 & 127) << 2;
    v4f v = *(const v4f*)(x + (size_t)(base + row) * DD + c4);
    v4h h;
#pragma unroll
    for (int j = 0; j < 4; ++j) h[j] = (_Float16)v[j];
    *(v4h*)(A + row * STRIDE + c4) = h;
  }
  __syncthreads();

  const int w      = tid >> 5;       // wave 0..7
  const int lane   = tid & 31;
  const int lane16 = lane & 15;
  const int sel    = lane >> 4;      // half-wave select for fragment striping
  const int mt     = w & 3;          // row tile 0..3 (16 rows each)
  const int g      = w >> 2;         // column half 0..1 (16 col tiles each)

  for (int l = 0; l < 3; ++l) {
    const _Float16* Wl  = W16 + (size_t)l * DD * DD;
    const float* bias   = (l == 0) ? b0 : (l == 1) ? b1 : bt;
    const _Float16* arow = A + (mt * 16 + lane16) * STRIDE;  // A-frag row for this lane

    for (int i = 0; i < 16; ++i) {
      const int ct = g * 16 + i;             // output column tile
      const int n  = ct * 16 + lane16;       // output column for this lane
      const float bv = bias[n];
      const _Float16* wrow = Wl + (size_t)n * DD;  // B column == W row (contiguous K)
      __builtin_prefetch(wrow, 0, 0);

      v8f acc = {};
#pragma unroll
      for (int kk = 0; kk < DD; kk += 32) {
        FragH af, bf;
        // A 16x32 f16 fragment: lane m / m+16 hold row m, K chunks {0..7,16..23} / {8..15,24..31}
        af.h[0] = *(const v8h*)(arow + kk + sel * 8);
        af.h[1] = *(const v8h*)(arow + kk + 16 + sel * 8);
        // B 32x16 f16 fragment: lane n / n+16 hold column n, K chunks {0..7,16..23} / {8..15,24..31}
        bf.h[0] = *(const v8h*)(wrow + kk + sel * 8);
        bf.h[1] = *(const v8h*)(wrow + kk + 16 + sel * 8);
        acc = __builtin_amdgcn_wmma_f32_16x16x32_f16(false, af.v, false, bf.v,
                                                     (short)0, acc, false, false);
      }

      if (l < 2) {
        // bias + tanh, store f16 into next layer's A buffer
#pragma unroll
        for (int j = 0; j < 8; ++j) {
          const int m = mt * 16 + sel * 8 + j;   // C layout: VGPR j, lane<16 -> M=j, else M=j+8
          H[m * STRIDE + n] = (_Float16)TANHF(acc[j] + bv);
        }
      } else {
        // head layer: bias only, write y to global as f32 (coalesced across lanes)
#pragma unroll
        for (int j = 0; j < 8; ++j) {
          const int m = mt * 16 + sel * 8 + j;
          out[(size_t)(base + m) * DD + n] = acc[j] + bv;
        }
      }
    }
    __syncthreads();

    if (l < 2) {
      // h_final: bands ending at s == S-1 copy their last row (h0 / h1)
      if ((base & 2047) == (2048 - MT)) {
        const int t = base + MT - 1;
        const int b = t >> 11;
        for (int d = tid; d < DD; d += 256)
          out[YTOT + (size_t)l * (8 * DD) + (size_t)b * DD + d] =
              (float)H[(MT - 1) * STRIDE + d];
      }
      _Float16* tmp = A; A = H; H = tmp;   // output becomes next layer's input
    }
  }
}

extern "C" void kernel_launch(void* const* d_in, const int* in_sizes, int n_in,
                              void* d_out, int out_size, void* d_ws, size_t ws_size,
                              hipStream_t stream) {
  const float* x  = (const float*)d_in[0];
  const float* W0 = (const float*)d_in[1];
  const float* b0 = (const float*)d_in[2];
  const float* W1 = (const float*)d_in[3];
  const float* b1 = (const float*)d_in[4];
  const float* Wt = (const float*)d_in[5];
  const float* bt = (const float*)d_in[6];
  float* out = (float*)d_out;

  _Float16* W16 = (_Float16*)d_ws;   // 3 * 512 * 512 * 2B = 1.5 MB scratch

  // 512*512/4 float4 per matrix = 65536 threads -> 256 blocks x 256 threads
  cvt_w_f16<<<(DD * DD / 4) / 256, 256, 0, stream>>>(W0, W1, Wt, W16);

  const int lds_bytes = 2 * MT * STRIDE * (int)sizeof(_Float16);  // 132 KB
  (void)hipFuncSetAttribute((const void*)mlp3_wmma,
                            hipFuncAttributeMaxDynamicSharedMemorySize, lds_bytes);

  mlp3_wmma<<<NTOK / MT, 256, lds_bytes, stream>>>(x, W16, b0, b1, bt, out);
}
